// RSNN_72430328479897
// MI455X (gfx1250) — matrix operand order repaired and checked by reading, not compiled, not run on hardware
//
#include <hip/hip_runtime.h>
#include <stdint.h>

#define BETA    0.9f
#define THRESH  1.0f
#define TSTEPS  20

#define B_TOT      16384
#define H_DIM      256
#define OUT_DIM    128
#define WMAIN_COLS 260

#define KT        8            // spike K tiles (256 / 32)
#define NT1       16           // H / 16 (global N tiles, recurrent GEMM)
#define NT2       8            // OUT / 16 (global N tiles, readout GEMM)
#define NW1       8            // N tiles per wave, recurrent (NT1 / 2 col-groups)
#define NW2       4            // N tiles per wave, readout  (NT2 / 2 col-groups)
#define EXTRA_KT  8            // index of the input/bias K-tile

#define FRAG_HALFS 512                         // 32 lanes * 16 halfs
#define WREC_FRAGS ((KT + 1) * NT1)            // 144
#define WOUT_FRAGS ((KT + 1) * NT2)            // 72
#define WREC_HALFS (WREC_FRAGS * FRAG_HALFS)
#define WGT_HALFS  ((WREC_FRAGS + WOUT_FRAGS) * FRAG_HALFS)
#define WGT_BYTES  (WGT_HALFS * 2)             // 221184
#define SROW_HALFS (16 * H_DIM)                // 4096 halfs = 8KB per 16-row tile
#define SSTAGE_BYTES (4 * 2 * SROW_HALFS * 2)  // 4 row-groups x 2 buffers = 65536
#define SMEM_BYTES (WGT_BYTES + SSTAGE_BYTES)  // 286720 <= 320KB WGP LDS

typedef __attribute__((ext_vector_type(16))) __bf16 v16bf;
typedef __attribute__((ext_vector_type(8)))  float  v8f;

union Frag { v16bf bf; uint4 q[2]; unsigned short us[16]; };

__device__ __forceinline__ unsigned short f2bf(float f) {
  union { float f; unsigned u; } c; c.f = f;
  unsigned u = c.u;
  return (unsigned short)((u + 0x7FFFu + ((u >> 16) & 1u)) >> 16); // RNE
}

// ---------------------------------------------------------------------------
// Prologue: pack W_main/W_out (+biases) into bf16 WMMA B-fragment layout.
// B-layout (ISA 7.12.2, 32x16 bf16): lane<16 -> K {0..7, 16..23}, lane>=16 -> K {8..15, 24..31},
// column N = 16*nt + (lane&15); each lane holds 16 halfs as two 8-K runs.
// Extra K-tile (kt==8): rows 0..3 = W_in columns, row 4 = bias, rest 0.
// ---------------------------------------------------------------------------
__global__ void rsnn_pack(const float* __restrict__ Wm, const float* __restrict__ bm,
                          const float* __restrict__ Wo, const float* __restrict__ bo,
                          unsigned short* __restrict__ ws) {
  int tid = blockIdx.x * blockDim.x + threadIdx.x;
  if (tid >= (WREC_FRAGS + WOUT_FRAGS) * 32) return;
  Frag f;
  unsigned short* dst;
  if (tid < WREC_FRAGS * 32) {
    int kt = tid >> 9;          // / (16 nt * 32 lanes)
    int nt = (tid >> 5) & 15;
    int lane = tid & 31;
    int n = nt * 16 + (lane & 15);
    #pragma unroll
    for (int j = 0; j < 16; ++j) {
      int klB = ((lane & 16) >> 1) + (j < 8 ? j : j + 8);
      float v;
      if (kt < KT)       v = Wm[n * WMAIN_COLS + 4 + kt * 32 + klB];
      else if (klB < 4)  v = Wm[n * WMAIN_COLS + klB];
      else if (klB == 4) v = bm[n];
      else               v = 0.0f;
      f.us[j] = f2bf(v);
    }
    dst = ws + (kt * NT1 + nt) * FRAG_HALFS + lane * 16;
  } else {
    int t2 = tid - WREC_FRAGS * 32;
    int kt = t2 >> 8;           // / (8 nt * 32 lanes)
    int nt = (t2 >> 5) & 7;
    int lane = t2 & 31;
    int n = nt * 16 + (lane & 15);
    #pragma unroll
    for (int j = 0; j < 16; ++j) {
      int klB = ((lane & 16) >> 1) + (j < 8 ? j : j + 8);
      float v;
      if (kt < KT)       v = Wo[n * H_DIM + kt * 32 + klB];
      else if (klB == 4) v = bo[n];
      else               v = 0.0f;
      f.us[j] = f2bf(v);
    }
    dst = ws + WREC_HALFS + (kt * NT2 + nt) * FRAG_HALFS + lane * 16;
  }
  uint4* q = (uint4*)dst;
  q[0] = f.q[0];
  q[1] = f.q[1];
}

// ---------------------------------------------------------------------------
// Main persistent RSNN kernel.
// 8 waves = 4 row-groups (16 batch rows each) x 2 column-groups (half of H).
// acc1 (mem_l): 8 tiles = 64 VGPRs/lane; acc2 (mem_o): 4 tiles = 32 VGPRs/lane.
// Spike staging double-buffered in LDS; one barrier per timestep.
// ---------------------------------------------------------------------------
#define WMMA_BF16(A, B, C) \
  __builtin_amdgcn_wmma_f32_16x16x32_bf16(false, (A), false, (B), (short)0, (C), false, false)

__device__ __forceinline__ v16bf load_bfrag(const unsigned short* lds_w, int frag, int lane) {
  const uint4* p = (const uint4*)(lds_w + frag * FRAG_HALFS + lane * 16);
  Frag f; f.q[0] = p[0]; f.q[1] = p[1];
  return f.bf;
}

// A-layout (ISA 7.12.2, 16x32 bf16): row M = lane&15;
// lane<16 -> K {0..7, 16..23}; lane>=16 -> K {8..15, 24..31}; each half 16B contiguous.
__device__ __forceinline__ v16bf load_afrag(const unsigned short* sb, int kt, int lane) {
  int h0 = (lane & 15) * H_DIM + kt * 32 + ((lane & 16) ? 8 : 0);
  Frag f;
  f.q[0] = *(const uint4*)(sb + h0);
  f.q[1] = *(const uint4*)(sb + h0 + 16);
  return f.bf;
}

__global__ void __launch_bounds__(256, 1)
rsnn_main(const float* __restrict__ inp, const unsigned short* __restrict__ wpack,
          float* __restrict__ out) {
  extern __shared__ unsigned char smem[];
  unsigned short* lds_w = (unsigned short*)smem;
  unsigned short* sstage = (unsigned short*)(smem + WGT_BYTES);

  const int tid  = threadIdx.x;
  const int lane = tid & 31;
  const int wave = tid >> 5;
  const int rowg = wave & 3;   // row-group: which 16 batch rows
  const int colg = wave >> 2;  // column-group: which half of H / half of OUT
  const int rowBase = blockIdx.x * 64 + rowg * 16;

  unsigned short* sbuf0 = sstage + (rowg * 2 + 0) * SROW_HALFS;
  unsigned short* sbuf1 = sstage + (rowg * 2 + 1) * SROW_HALFS;

  // ---- stage packed weights into LDS via async global->LDS DMA ----
  {
    unsigned gss =
#if __has_builtin(__builtin_amdgcn_groupstaticsize)
        __builtin_amdgcn_groupstaticsize();
#else
        0u;
#endif
    const unsigned char* g = (const unsigned char*)wpack;
    #pragma unroll 1
    for (int i = 0; i < WGT_BYTES / (16 * 256); ++i) {   // 54 iters, 16B/lane each
      unsigned off = (unsigned)(tid + i * 256) * 16u;
      unsigned lds_addr = gss + off;
      uint64_t gaddr = (uint64_t)(uintptr_t)(g + off);
      asm volatile("global_load_async_to_lds_b128 %0, %1, off"
                   :: "v"(lds_addr), "v"(gaddr)
                   : "memory");
    }
#if __has_builtin(__builtin_amdgcn_s_wait_asynccnt)
    __builtin_amdgcn_s_wait_asynccnt(0);
#else
    asm volatile("s_wait_asynccnt 0" ::: "memory");
#endif
  }
  __syncthreads();

  // ---- constant A fragment: cols 0..3 = inp row, col 4 = 1.0 (bias lane) ----
  Frag fe;
  {
    int r = rowBase + (lane & 15);
    #pragma unroll
    for (int j = 0; j < 16; ++j) {
      int kl = ((lane & 16) ? 8 : 0) + (j < 8 ? j : j + 8);
      float v = (kl < 4) ? inp[r * 4 + kl] : (kl == 4 ? 1.0f : 0.0f);
      fe.us[j] = f2bf(v);
    }
  }
  const v16bf a_ext = fe.bf;

  const v8f z8 = {0.f, 0.f, 0.f, 0.f, 0.f, 0.f, 0.f, 0.f};
  v8f acc1[NW1];   // mem_l slice, resident in accumulator registers
  v8f acc2[NW2];   // mem_o slice
  #pragma unroll
  for (int i = 0; i < NW1; ++i) acc1[i] = z8;
  #pragma unroll
  for (int i = 0; i < NW2; ++i) acc2[i] = z8;

  const int m0  = (lane >> 4) * 8;   // D-layout: VGPR e -> row m0+e
  const int nlo = lane & 15;

  // ---- t = 0: mem0 = inp@Win + b (extra tile only); s0 -> buf0; membrane discarded
  #pragma unroll
  for (int j = 0; j < NW1; ++j) {
    int ntg = colg * NW1 + j;
    v16bf b = load_bfrag(lds_w, EXTRA_KT * NT1 + ntg, lane);
    v8f m = WMMA_BF16(a_ext, b, z8);
    #pragma unroll
    for (int e = 0; e < 8; ++e) {
      float sp = (m[e] > THRESH) ? 1.0f : 0.0f;
      sbuf0[(m0 + e) * H_DIM + ntg * 16 + nlo] =
          (sp != 0.0f) ? (unsigned short)0x3F80 : (unsigned short)0;
    }
  }
  __syncthreads();

  // ---- steps t = 1..19 ----
  #pragma unroll 1
  for (int it = 0; it < TSTEPS - 1; ++it) {
    unsigned short* sc = (it & 1) ? sbuf1 : sbuf0;   // s_prev
    unsigned short* sn = (it & 1) ? sbuf0 : sbuf1;   // s_new
    // mem_l = beta*mem_l + [inp,1,s_prev] @ W_extended^T   (this wave's H slice)
    #pragma unroll
    for (int j = 0; j < NW1; ++j) {
      acc1[j] *= BETA;
      v16bf b = load_bfrag(lds_w, EXTRA_KT * NT1 + colg * NW1 + j, lane);
      acc1[j] = WMMA_BF16(a_ext, b, acc1[j]);
    }
    #pragma unroll
    for (int kt = 0; kt < KT; ++kt) {
      v16bf a = load_afrag(sc, kt, lane);
      #pragma unroll
      for (int j = 0; j < NW1; ++j) {
        v16bf b = load_bfrag(lds_w, kt * NT1 + colg * NW1 + j, lane);
        acc1[j] = WMMA_BF16(a, b, acc1[j]);
      }
    }
    // spike + soft reset, stage s_new into the other buffer
    #pragma unroll
    for (int j = 0; j < NW1; ++j) {
      int ntg = colg * NW1 + j;
      #pragma unroll
      for (int e = 0; e < 8; ++e) {
        float m = acc1[j][e];
        float sp = (m > THRESH) ? 1.0f : 0.0f;
        acc1[j][e] = m - sp;  // THRESH == 1
        sn[(m0 + e) * H_DIM + ntg * 16 + nlo] =
            (sp != 0.0f) ? (unsigned short)0x3F80 : (unsigned short)0;
      }
    }
    __syncthreads();   // publish s_new; also retires all reads of sc
    // mem_o = beta*mem_o + s_new @ W_out^T + b_out   (this wave's OUT slice)
    #pragma unroll
    for (int j = 0; j < NW2; ++j) {
      acc2[j] *= BETA;
      v16bf b = load_bfrag(lds_w, WREC_FRAGS + EXTRA_KT * NT2 + colg * NW2 + j, lane);
      acc2[j] = WMMA_BF16(a_ext, b, acc2[j]);
    }
    #pragma unroll
    for (int kt = 0; kt < KT; ++kt) {
      v16bf a = load_afrag(sn, kt, lane);
      #pragma unroll
      for (int j = 0; j < NW2; ++j) {
        v16bf b = load_bfrag(lds_w, WREC_FRAGS + kt * NT2 + colg * NW2 + j, lane);
        acc2[j] = WMMA_BF16(a, b, acc2[j]);
      }
    }
  }

  // ---- write mem_o (f32, row-major B x 128) ----
  #pragma unroll
  for (int j = 0; j < NW2; ++j) {
    int ntg = colg * NW2 + j;
    #pragma unroll
    for (int e = 0; e < 8; ++e) {
      out[(size_t)(rowBase + m0 + e) * OUT_DIM + ntg * 16 + nlo] = acc2[j][e];
    }
  }
}

// ---------------------------------------------------------------------------
extern "C" void kernel_launch(void* const* d_in, const int* in_sizes, int n_in,
                              void* d_out, int out_size, void* d_ws, size_t ws_size,
                              hipStream_t stream) {
  const float* inp    = (const float*)d_in[0];
  const float* W_main = (const float*)d_in[1];
  const float* b_main = (const float*)d_in[2];
  const float* W_out  = (const float*)d_in[3];
  const float* b_out  = (const float*)d_in[4];
  unsigned short* wpack = (unsigned short*)d_ws;   // needs 221184 bytes

  // pack weights -> bf16 WMMA fragments
  int packThreads = (WREC_FRAGS + WOUT_FRAGS) * 32;           // 6912
  rsnn_pack<<<(packThreads + 255) / 256, 256, 0, stream>>>(W_main, b_main,
                                                           W_out, b_out, wpack);

  // persistent RSNN: 64 rows/block, 8 waves (4 row-groups x 2 col-groups), 287KB LDS
  (void)hipFuncSetAttribute((const void*)rsnn_main,
                            hipFuncAttributeMaxDynamicSharedMemorySize, SMEM_BYTES);
  rsnn_main<<<B_TOT / 64, 256, SMEM_BYTES, stream>>>(inp, wpack, (float*)d_out);
}
